// DeepseekV3MoE_2937757630476
// MI455X (gfx1250) — matrix operand order repaired and checked by reading, not compiled, not run on hardware
//
#include <hip/hip_runtime.h>
#include <cstdint>
#include <cstddef>

// ---------------- types for WMMA ----------------
typedef __attribute__((ext_vector_type(16))) __bf16         v16bf;
typedef __attribute__((ext_vector_type(8)))  float          v8f;
typedef __attribute__((ext_vector_type(8)))  unsigned short v8us;

// ---------------- problem config ----------------
constexpr int   NTOK  = 2 * 2048;   // B*S
constexpr int   HD    = 1024;       // hidden
constexpr int   NE    = 16;         // experts
constexpr int   ID    = 512;        // moe intermediate
constexpr int   SHI   = 1024;       // shared expert intermediate
constexpr int   TOPK  = 4;
constexpr float RSCALE = 2.5f;
constexpr int   LDP   = 32;         // LDS tile row pitch (elements)

// ---------------- helpers ----------------
static __device__ __forceinline__ unsigned short f2bf(float f) {
  unsigned u = __builtin_bit_cast(unsigned, f);
  u += 0x7FFFu + ((u >> 16) & 1u);           // round-to-nearest-even
  return (unsigned short)(u >> 16);
}

union FragU { v16bf v; v8us h[2]; };
static __device__ __forceinline__ v16bf ld_frag(const unsigned short* p0,
                                                const unsigned short* p1) {
  FragU u;
  u.h[0] = *reinterpret_cast<const v8us*>(p0);
  u.h[1] = *reinterpret_cast<const v8us*>(p1);
  return u.v;
}

// ---- CDNA5 async global->LDS DMA (ASYNCcnt-tracked), per-lane 16B ----
// LDS flat pointers carry the wave-relative LDS offset in addr[31:0]
// (ISA 10.2 aperture mapping), so the low dword is the VDST LDS address.
static __device__ __forceinline__ void async_copy_b128(void* lds_dst,
                                                       const void* gsrc) {
  unsigned loff = (unsigned)(uintptr_t)lds_dst;
  asm volatile("global_load_async_to_lds_b128 %0, %1, off"
               :
               : "v"(loff), "v"(gsrc)
               : "memory");
}

#if defined(__has_builtin)
#if __has_builtin(__builtin_amdgcn_s_wait_asynccnt)
#define WAIT_ASYNC(n) __builtin_amdgcn_s_wait_asynccnt((short)(n))
#endif
#if __has_builtin(__builtin_amdgcn_s_wait_tensorcnt)
#define WAIT_TENSOR(n) __builtin_amdgcn_s_wait_tensorcnt((short)(n))
#endif
#if __has_builtin(__builtin_amdgcn_tensor_load_to_lds)
#define HAVE_TDM 1
#endif
#endif
#ifndef WAIT_ASYNC
#define WAIT_ASYNC(n) asm volatile("s_wait_asynccnt %0" ::"i"(n) : "memory")
#endif
#ifndef WAIT_TENSOR
#define WAIT_TENSOR(n) asm volatile("s_wait_tensorcnt %0" ::"i"(n) : "memory")
#endif

#ifdef HAVE_TDM
// ---- Tensor Data Mover: 2D tile load (128 rows x 32 bf16 cols) ----
// D# per CDNA5 ISA ch.8: group0 {count, lds_addr, global_addr, type=2},
// group1 {data_size=2B, tensor_dim0=32, tensor_dim1=128, tile 32x128,
//         dim0_stride}, groups 2/3 zero (2D tensor). TENSORcnt-tracked.
typedef __attribute__((ext_vector_type(4))) unsigned int v4u;
typedef __attribute__((ext_vector_type(8))) int          v8i;
typedef __attribute__((ext_vector_type(4))) int          v4i;

static __device__ __forceinline__ void tdm_load_tile(void* lds_dst,
                                                     const void* gsrc,
                                                     int stride_elems) {
  unsigned           lds_addr = (unsigned)(uintptr_t)lds_dst;
  unsigned long long ga       = (unsigned long long)(uintptr_t)gsrc;
  v4u g0;
  g0[0] = 1u;                                    // count=1 (valid user D#)
  g0[1] = lds_addr;                              // LDS byte address
  g0[2] = (unsigned)ga;                          // global_addr[95:64]
  g0[3] = (unsigned)((ga >> 32) & 0x1FFFFFFull)  // global_addr[120:96]
          | (2u << 30);                          // type=2 ("image")
  v8i g1;
  g1[0] = 1 << 16;                               // wg_mask=0, data_size=1(2B)
  g1[1] = (int)(32u << 16);                      // tensor_dim0=32 (lo16)
  g1[2] = (int)(128u << 16);                     // dim0 hi=0 | tensor_dim1=128
  g1[3] = (int)(32u << 16);                      // dim1 hi=0 | tile_dim0=32
  g1[4] = 128;                                   // tile_dim1=128, tile_dim2=0
  g1[5] = stride_elems;                          // tensor_dim0_stride lo32
  g1[6] = 0;                                     // stride hi, dim1_stride lo
  g1[7] = 0;                                     // dim1_stride hi
  v4i z4 = {0, 0, 0, 0};
#if defined(__clang_major__) && __clang_major__ >= 23
  v8i z8 = {0, 0, 0, 0, 0, 0, 0, 0};
  __builtin_amdgcn_tensor_load_to_lds(g0, g1, z4, z4, z8, 0);
#else
  __builtin_amdgcn_tensor_load_to_lds(g0, g1, z4, z4, 0);
#endif
}
#endif  // HAVE_TDM

// ---------------- small utility kernels ----------------
__global__ void cvt_f32_to_bf16(const float* __restrict__ s,
                                unsigned short* __restrict__ d, int n) {
  for (int i = blockIdx.x * blockDim.x + threadIdx.x; i < n;
       i += gridDim.x * blockDim.x)
    d[i] = f2bf(s[i]);
}

__global__ void zero_f32(float* __restrict__ d, int n) {
  for (int i = blockIdx.x * blockDim.x + threadIdx.x; i < n;
       i += gridDim.x * blockDim.x)
    d[i] = 0.0f;
}

__global__ void zero_counters(int* cnt, int* cur) {
  int t = threadIdx.x;
  if (t < NE) { cnt[t] = 0; cur[t] = 0; }
}

__global__ void count_kernel(const float* __restrict__ cw, int* __restrict__ cnt) {
  for (int n = blockIdx.x * blockDim.x + threadIdx.x; n < NTOK;
       n += gridDim.x * blockDim.x) {
#pragma unroll
    for (int e = 0; e < NE; ++e)
      if (cw[n * NE + e] > 0.0f) atomicAdd(&cnt[e], 1);
  }
}

__global__ void scan_kernel(const int* __restrict__ cnt, int* __restrict__ off) {
  if (threadIdx.x == 0 && blockIdx.x == 0) {
    int acc = 0;
#pragma unroll
    for (int e = 0; e < NE; ++e) { off[e] = acc; acc += cnt[e]; }
  }
}

__global__ void scatter_kernel(const float* __restrict__ cw,
                               const int* __restrict__ off, int* __restrict__ cur,
                               int* __restrict__ tok_list, float* __restrict__ wslot) {
  for (int n = blockIdx.x * blockDim.x + threadIdx.x; n < NTOK;
       n += gridDim.x * blockDim.x) {
#pragma unroll
    for (int e = 0; e < NE; ++e) {
      float w = cw[n * NE + e];
      if (w > 0.0f) {
        int p = atomicAdd(&cur[e], 1);
        int s = off[e] + p;
        tok_list[s] = n;
        wslot[s]    = w;
      }
    }
  }
}

__global__ void final_add(const float* __restrict__ x, const float* __restrict__ yacc,
                          float* __restrict__ out, int n) {
  for (int i = blockIdx.x * blockDim.x + threadIdx.x; i < n;
       i += gridDim.x * blockDim.x)
    out[i] = x[i] + yacc[i];
}

// ---------------- gate / routing (fp32, VALU) ----------------
__global__ void gate_topk(const float* __restrict__ x, const float* __restrict__ gw,
                          const float* __restrict__ gb, float* __restrict__ cw) {
  int n = blockIdx.x * blockDim.x + threadIdx.x;
  if (n >= NTOK) return;

  float acc[NE];
#pragma unroll
  for (int e = 0; e < NE; ++e) acc[e] = 0.0f;
  const float* xr = x + (size_t)n * HD;
  for (int h = 0; h < HD; ++h) {
    float xv = xr[h];
#pragma unroll
    for (int e = 0; e < NE; ++e) acc[e] += xv * gw[e * HD + h];
  }

  float scores[NE], sfc[NE];
#pragma unroll
  for (int e = 0; e < NE; ++e) {
    scores[e] = 1.0f / (1.0f + expf(-acc[e]));
    sfc[e]    = scores[e] + gb[e];
  }

  // group score = sum of top-2 within each group of 4
  float gsum[4];
#pragma unroll
  for (int g = 0; g < 4; ++g) {
    float m1 = -1e30f, m2 = -1e30f;
#pragma unroll
    for (int i = 0; i < 4; ++i) {
      float v = sfc[g * 4 + i];
      if (v > m1) { m2 = m1; m1 = v; } else if (v > m2) { m2 = v; }
    }
    gsum[g] = m1 + m2;
  }
  // top-2 groups (stable: strict > keeps lowest index on ties)
  int g0 = 0;
#pragma unroll
  for (int g = 1; g < 4; ++g) if (gsum[g] > gsum[g0]) g0 = g;
  int g1 = -1;
#pragma unroll
  for (int g = 0; g < 4; ++g)
    if (g != g0 && (g1 < 0 || gsum[g] > gsum[g1])) g1 = g;

  bool used[NE];
#pragma unroll
  for (int e = 0; e < NE; ++e) used[e] = false;

  int   idx[TOPK];
  float w[TOPK];
#pragma unroll
  for (int r = 0; r < TOPK; ++r) {
    float best = -1e30f; int bi = -1;
#pragma unroll
    for (int e = 0; e < NE; ++e) {
      int grp = e >> 2;
      if ((grp == g0 || grp == g1) && !used[e] && sfc[e] > best) {
        best = sfc[e]; bi = e;
      }
    }
    used[bi] = true;
    idx[r]   = bi;
    w[r]     = scores[bi];
  }
  float s = w[0] + w[1] + w[2] + w[3] + 1e-20f;
  float sc = RSCALE / s;

  float row[NE];
#pragma unroll
  for (int e = 0; e < NE; ++e) row[e] = 0.0f;
#pragma unroll
  for (int r = 0; r < TOPK; ++r) row[idx[r]] = w[r] * sc;
#pragma unroll
  for (int e = 0; e < NE; ++e) cw[n * NE + e] = row[e];
}

// ---------------- GEMM 1: act = silu(X*Wg^T) * (X*Wu^T) * rowScale ----------------
// Tile 128x128, 8 waves, bf16 WMMA 16x16x32, fp32 accumulate.
// X tile: per-lane GLOBAL_LOAD_ASYNC_TO_LDS_B128 (gathered rows, ASYNCcnt).
// Weight tiles: TDM tensor_load_to_lds (TENSORcnt), one descriptor per tile.
__global__ __launch_bounds__(256) void gemm_gateup(
    const unsigned short* __restrict__ xbf,
    const unsigned short* __restrict__ Wg,
    const unsigned short* __restrict__ Wu,
    unsigned short* __restrict__ act,
    const int*   __restrict__ tokmap,  // null => identity rows
    const int*   __restrict__ cnt,     // null => totalRows
    const int*   __restrict__ off,     // null => 0
    const float* __restrict__ wslot,   // null => 1.0
    int totalRows, int Ncols, int Kdim) {

  const int e     = blockIdx.z;
  const int count = cnt ? cnt[e] : totalRows;
  const int base  = off ? off[e] : 0;
  const int mbase = blockIdx.y * 128;
  if (mbase >= count) return;
  const int nbase = blockIdx.x * 128;

  const unsigned short* Wge = Wg + (size_t)e * Ncols * Kdim;
  const unsigned short* Wue = Wu + (size_t)e * Ncols * Kdim;

  __shared__ unsigned short Xs[2][128 * LDP];   // 2 x 8KB
  __shared__ unsigned short Wgs[2][128 * LDP];  // 2 x 8KB
  __shared__ unsigned short Wus[2][128 * LDP];  // 2 x 8KB
  __shared__ int   toksS[128];
  __shared__ float sclS[128];

  const int tid = threadIdx.x;
  if (tid < 128) {
    int row = mbase + tid;
    int t = (row < count) ? (tokmap ? tokmap[base + row] : row) : 0;
    toksS[tid] = t;
    sclS[tid]  = (wslot && row < count) ? wslot[base + row] : 1.0f;
  }
  __syncthreads();

  const int lane = tid & 31;
  const int wv   = tid >> 5;

  // gathered X tile: 2 async b128 copies per thread
  auto stageX = [&](int k0, int p) {
#pragma unroll
    for (int s2 = 0; s2 < 2; ++s2) {
      int s = tid + s2 * 256;
      int r = s >> 2, c = (s & 3) * 8;
      async_copy_b128(&Xs[p][r * LDP + c],
                      &xbf[(size_t)toksS[r] * Kdim + k0 + c]);
    }
  };
  // weight tiles
  auto stageW = [&](int k0, int p) {
#ifdef HAVE_TDM
    if (wv == 0)
      tdm_load_tile(&Wgs[p][0], &Wge[(size_t)nbase * Kdim + k0], Kdim);
    else if (wv == 1)
      tdm_load_tile(&Wus[p][0], &Wue[(size_t)nbase * Kdim + k0], Kdim);
#else
#pragma unroll
    for (int s2 = 0; s2 < 2; ++s2) {
      int s = tid + s2 * 256;
      int r = s >> 2, c = (s & 3) * 8;
      async_copy_b128(&Wgs[p][r * LDP + c],
                      &Wge[(size_t)(nbase + r) * Kdim + k0 + c]);
      async_copy_b128(&Wus[p][r * LDP + c],
                      &Wue[(size_t)(nbase + r) * Kdim + k0 + c]);
    }
#endif
  };

  v8f cg[8], cu[8];
  const v8f z8 = {0.f, 0.f, 0.f, 0.f, 0.f, 0.f, 0.f, 0.f};
#pragma unroll
  for (int j = 0; j < 8; ++j) { cg[j] = z8; cu[j] = z8; }

  const int T = Kdim / 32;
  stageX(0, 0);                    // prologue: tile 0 -> buffer 0
  stageW(0, 0);

  for (int kt = 0; kt < T; ++kt) {
    const int p = kt & 1;
    if (kt + 1 < T) {
      stageX((kt + 1) * 32, p ^ 1);  // prefetch next tile into other buffer
      stageW((kt + 1) * 32, p ^ 1);
#ifdef HAVE_TDM
      WAIT_ASYNC(2);                 // in-order => X tile kt landed
      if (wv < 2) WAIT_TENSOR(1);    // in-order => weight tile kt landed
#else
      WAIT_ASYNC(6);
#endif
    } else {
      WAIT_ASYNC(0);
#ifdef HAVE_TDM
      if (wv < 2) WAIT_TENSOR(0);
#endif
    }
    __syncthreads();

    const int mrow = wv * 16 + (lane & 15);
    const int kb   = (lane >> 4) * 8;     // A: k-pair base within 16-half
    const int kh   = (lane >> 4) * 16;    // unused; B is loaded with kb, like A
    const unsigned short* Xb = Xs[p];
    const unsigned short* Gb = Wgs[p];
    const unsigned short* Ub = Wus[p];
    v16bf A = ld_frag(&Xb[mrow * LDP + kb], &Xb[mrow * LDP + 16 + kb]);

#pragma unroll
    for (int j = 0; j < 8; ++j) {
      int nr = j * 16 + (lane & 15);
      v16bf Bg = ld_frag(&Gb[nr * LDP + kb], &Gb[nr * LDP + 16 + kb]);
      cg[j] = __builtin_amdgcn_wmma_f32_16x16x32_bf16(
          false, A, false, Bg, (short)0, cg[j], false, false);
      v16bf Bu = ld_frag(&Ub[nr * LDP + kb], &Ub[nr * LDP + 16 + kb]);
      cu[j] = __builtin_amdgcn_wmma_f32_16x16x32_bf16(
          false, A, false, Bu, (short)0, cu[j], false, false);
    }
    __syncthreads();
  }

  // epilogue: silu(g) * u * scale, store bf16
#pragma unroll
  for (int j = 0; j < 8; ++j) {
    int col = nbase + j * 16 + (lane & 15);
#pragma unroll
    for (int v = 0; v < 8; ++v) {
      int rloc = wv * 16 + (lane >> 4) * 8 + v;
      int row  = mbase + rloc;
      if (row < count) {
        float g = cg[j][v];
        float u = cu[j][v];
        float a = (g / (1.0f + expf(-g))) * u * sclS[rloc];
        act[(size_t)(base + row) * Ncols + col] = f2bf(a);
      }
    }
  }
}

// ---------------- GEMM 2: yacc[tok, :] += act_row @ W^T (atomic scatter) ----------------
// Both tiles (act rows are contiguous slots, W rows contiguous) are plain 2D
// tiles -> both staged by TDM when available.
__global__ __launch_bounds__(256) void gemm_down(
    float* __restrict__ yacc,
    const unsigned short* __restrict__ act,
    const unsigned short* __restrict__ W,   // [HD, Kdim] per expert
    const int* __restrict__ tokmap,
    const int* __restrict__ cnt,
    const int* __restrict__ off,
    int totalRows, int Kdim) {

  const int e     = blockIdx.z;
  const int count = cnt ? cnt[e] : totalRows;
  const int base  = off ? off[e] : 0;
  const int mbase = blockIdx.y * 128;
  if (mbase >= count) return;
  const int nbase = blockIdx.x * 128;

  const unsigned short* We = W + (size_t)e * HD * Kdim;

  __shared__ unsigned short As[2][128 * LDP];
  __shared__ unsigned short Ws[2][128 * LDP];
  __shared__ int toksS[128];

  const int tid = threadIdx.x;
  if (tid < 128) {
    int row = mbase + tid;
    toksS[tid] = (row < count) ? (tokmap ? tokmap[base + row] : row) : 0;
  }
  __syncthreads();

  const int lane = tid & 31;
  const int wv   = tid >> 5;

  auto stage = [&](int k0, int p) {
#ifdef HAVE_TDM
    if (wv == 0)
      tdm_load_tile(&Ws[p][0], &We[(size_t)nbase * Kdim + k0], Kdim);
    else if (wv == 1)
      tdm_load_tile(&As[p][0],
                    &act[(size_t)(base + mbase) * Kdim + k0], Kdim);
#else
#pragma unroll
    for (int s2 = 0; s2 < 2; ++s2) {
      int s = tid + s2 * 256;
      int r = s >> 2, c = (s & 3) * 8;
      async_copy_b128(&As[p][r * LDP + c],
                      &act[(size_t)(base + mbase + r) * Kdim + k0 + c]);
      async_copy_b128(&Ws[p][r * LDP + c],
                      &We[(size_t)(nbase + r) * Kdim + k0 + c]);
    }
#endif
  };

  v8f c[8];
  const v8f z8 = {0.f, 0.f, 0.f, 0.f, 0.f, 0.f, 0.f, 0.f};
#pragma unroll
  for (int j = 0; j < 8; ++j) c[j] = z8;

  const int T = Kdim / 32;
  stage(0, 0);

  for (int kt = 0; kt < T; ++kt) {
    const int p = kt & 1;
    if (kt + 1 < T) {
      stage((kt + 1) * 32, p ^ 1);
#ifdef HAVE_TDM
      if (wv < 2) WAIT_TENSOR(1);
#else
      WAIT_ASYNC(4);
#endif
    } else {
#ifdef HAVE_TDM
      if (wv < 2) WAIT_TENSOR(0);
#else
      WAIT_ASYNC(0);
#endif
    }
    __syncthreads();

    const int mrow = wv * 16 + (lane & 15);
    const int kb   = (lane >> 4) * 8;
    const int kh   = (lane >> 4) * 16;
    const unsigned short* Ab = As[p];
    const unsigned short* Wb = Ws[p];
    v16bf A = ld_frag(&Ab[mrow * LDP + kb], &Ab[mrow * LDP + 16 + kb]);

#pragma unroll
    for (int j = 0; j < 8; ++j) {
      int nr = j * 16 + (lane & 15);
      v16bf B = ld_frag(&Wb[nr * LDP + kb], &Wb[nr * LDP + 16 + kb]);
      c[j] = __builtin_amdgcn_wmma_f32_16x16x32_bf16(
          false, A, false, B, (short)0, c[j], false, false);
    }
    __syncthreads();
  }

#pragma unroll
  for (int j = 0; j < 8; ++j) {
    int col = nbase + j * 16 + (lane & 15);
#pragma unroll
    for (int v = 0; v < 8; ++v) {
      int rloc = wv * 16 + (lane >> 4) * 8 + v;
      int row  = mbase + rloc;
      if (row < count) {
        atomicAdd(&yacc[(size_t)toksS[rloc] * HD + col], c[j][v]);
      }
    }
  }
}

// ---------------- host launcher ----------------
extern "C" void kernel_launch(void* const* d_in, const int* in_sizes, int n_in,
                              void* d_out, int out_size, void* d_ws, size_t ws_size,
                              hipStream_t stream) {
  const float* x    = (const float*)d_in[0];  // [N, HD]
  const float* gw   = (const float*)d_in[1];  // [NE, HD]
  const float* gb   = (const float*)d_in[2];  // [NE]
  const float* Wg   = (const float*)d_in[3];  // [NE, ID, HD]
  const float* Wu   = (const float*)d_in[4];  // [NE, ID, HD]
  const float* Wd   = (const float*)d_in[5];  // [NE, HD, ID]
  const float* Wsg  = (const float*)d_in[6];  // [SHI, HD]
  const float* Wsu  = (const float*)d_in[7];  // [SHI, HD]
  const float* Wsd  = (const float*)d_in[8];  // [HD, SHI]
  float* out = (float*)d_out;

  unsigned char* p = (unsigned char*)d_ws;
  auto carve = [&](size_t bytes) -> void* {
    void* r = (void*)p;
    p += (bytes + 255) & ~(size_t)255;
    return r;
  };

  const int NSLOT = NTOK * TOPK;
  unsigned short* xbf   = (unsigned short*)carve((size_t)NTOK * HD * 2);
  unsigned short* WgB   = (unsigned short*)carve((size_t)NE * ID * HD * 2);
  unsigned short* WuB   = (unsigned short*)carve((size_t)NE * ID * HD * 2);
  unsigned short* WdB   = (unsigned short*)carve((size_t)NE * HD * ID * 2);
  unsigned short* WsgB  = (unsigned short*)carve((size_t)SHI * HD * 2);
  unsigned short* WsuB  = (unsigned short*)carve((size_t)SHI * HD * 2);
  unsigned short* WsdB  = (unsigned short*)carve((size_t)HD * SHI * 2);
  float* cw             = (float*)carve((size_t)NTOK * NE * 4);
  int*   cnt            = (int*)carve(NE * 4);
  int*   cur            = (int*)carve(NE * 4);
  int*   off            = (int*)carve(NE * 4);
  int*   tok_list       = (int*)carve((size_t)(NSLOT + 128) * 4);
  float* wslot          = (float*)carve((size_t)(NSLOT + 128) * 4);
  unsigned short* actR  = (unsigned short*)carve((size_t)(NSLOT + 128) * ID * 2);
  unsigned short* actS  = (unsigned short*)carve((size_t)NTOK * SHI * 2);
  float* yacc           = (float*)carve((size_t)NTOK * HD * 4);
  (void)ws_size; (void)n_in; (void)in_sizes; (void)out_size;

  // 1) bf16 conversions
  cvt_f32_to_bf16<<<2048, 256, 0, stream>>>(x,   xbf,  NTOK * HD);
  cvt_f32_to_bf16<<<4096, 256, 0, stream>>>(Wg,  WgB,  NE * ID * HD);
  cvt_f32_to_bf16<<<4096, 256, 0, stream>>>(Wu,  WuB,  NE * ID * HD);
  cvt_f32_to_bf16<<<4096, 256, 0, stream>>>(Wd,  WdB,  NE * HD * ID);
  cvt_f32_to_bf16<<<1024, 256, 0, stream>>>(Wsg, WsgB, SHI * HD);
  cvt_f32_to_bf16<<<1024, 256, 0, stream>>>(Wsu, WsuB, SHI * HD);
  cvt_f32_to_bf16<<<1024, 256, 0, stream>>>(Wsd, WsdB, HD * SHI);

  // 2) routing
  gate_topk<<<NTOK / 128, 128, 0, stream>>>(x, gw, gb, cw);
  zero_counters<<<1, 32, 0, stream>>>(cnt, cur);
  count_kernel<<<64, 256, 0, stream>>>(cw, cnt);
  scan_kernel<<<1, 1, 0, stream>>>(cnt, off);
  scatter_kernel<<<64, 256, 0, stream>>>(cw, off, cur, tok_list, wslot);

  // 3) output accumulator
  zero_f32<<<4096, 256, 0, stream>>>(yacc, NTOK * HD);

  // 4) routed experts: gate/up -> act (weighted), then down -> yacc (atomic)
  gemm_gateup<<<dim3(ID / 128, NTOK / 128, NE), 256, 0, stream>>>(
      xbf, WgB, WuB, actR, tok_list, cnt, off, wslot, 0, ID, HD);
  gemm_down<<<dim3(HD / 128, NTOK / 128, NE), 256, 0, stream>>>(
      yacc, actR, WdB, tok_list, cnt, off, 0, ID);

  // 5) shared expert (identity token map, full N rows, no scaling)
  gemm_gateup<<<dim3(SHI / 128, NTOK / 128, 1), 256, 0, stream>>>(
      xbf, WsgB, WsuB, actS, nullptr, nullptr, nullptr, nullptr, NTOK, SHI, HD);
  gemm_down<<<dim3(HD / 128, NTOK / 128, 1), 256, 0, stream>>>(
      yacc, actS, WsdB, nullptr, nullptr, nullptr, NTOK, SHI);

  // 6) residual add
  final_add<<<4096, 256, 0, stream>>>(x, yacc, out, NTOK * HD);
}
